// ContractiveMessageBlock_1451698946762
// MI455X (gfx1250) — matrix operand order repaired and checked by reading, not compiled, not run on hardware
//
#include <hip/hip_runtime.h>

#define FEAT     128
#define N_RBF    20
#define N_EDGES  200000
#define N_CG     2000
#define PI_F     3.14159265358979f
#define CUTOFF_F 5.0f
#define EPB      32                       // edges per block

typedef __attribute__((ext_vector_type(16))) __bf16 v16bf;
typedef __attribute__((ext_vector_type(8)))  float  v8f;

__device__ __forceinline__ unsigned short f2bf(float f) {
    unsigned int u = __float_as_uint(f);
    u += 0x7FFFu + ((u >> 16) & 1u);       // round-to-nearest-even bf16
    return (unsigned short)(u >> 16);
}

// Low 32 bits of a generic pointer to LDS == LDS byte address (aperture truncation).
__device__ __forceinline__ unsigned lds_addr_of(const void* p) {
    return (unsigned)(uintptr_t)p;
}

// CDNA5 async global->LDS copy, 16B per active lane. Tracked by ASYNCcnt.
__device__ __forceinline__ void async_copy16(unsigned lds_byte_addr, const void* gptr) {
    asm volatile("global_load_async_to_lds_b128 %0, %1, off"
                 :: "v"(lds_byte_addr), "v"((unsigned long long)(uintptr_t)gptr)
                 : "memory");
}

__device__ __forceinline__ void wait_async0() {
    asm volatile("s_wait_asynccnt 0" ::: "memory");
}

// Gather WMMA A-operand (16x32 bf16 tile) from an LDS row-major 16x128 bf16 buffer.
// ISA layout: lanes 0-15 rows M=lane, lanes 16-31 rows M=lane-16;
// VGPR i holds K pair: K = 16*(i/4) + 8*(lane/16) + 2*(i%4).
__device__ __forceinline__ v16bf gather_a(const unsigned short* lds, int lane, int kb) {
    union { v16bf v; unsigned int w[8]; } a;
    const int m     = lane & 15;
    const int klane = (lane >> 4) << 3;
    const unsigned short* row = lds + m * FEAT + (kb << 5);
#pragma unroll
    for (int i = 0; i < 8; ++i) {
        const int K = ((i >> 2) << 4) + klane + ((i & 3) << 1);
        a.w[i] = *(const unsigned int*)(row + K);   // 2 consecutive bf16, 4B aligned
    }
    return a.v;
}

// Load WMMA B-operand (32x16 bf16 tile) pre-packed so each lane's 16 values are contiguous.
__device__ __forceinline__ v16bf load_b(const unsigned short* gp) {
    union { v16bf v; unsigned int w[8]; } b;
    const unsigned int* q = (const unsigned int*)gp;
#pragma unroll
    for (int i = 0; i < 8; ++i) b.w[i] = q[i];
    return b.v;
}

// ---------------- auxiliary kernels ----------------

__global__ void zero_kernel(float* out, int n, float* counts, int nc) {
    const int stride = gridDim.x * blockDim.x;
    for (int i = blockIdx.x * blockDim.x + threadIdx.x; i < n;  i += stride) out[i]    = 0.0f;
    for (int j = blockIdx.x * blockDim.x + threadIdx.x; j < nc; j += stride) counts[j] = 0.0f;
}

// Pack W (KDIM x NDIM f32, row-major) into bf16 WMMA-B tile layout:
// dst[((kt*ntiles + nt)*32 + lane)*16 + j] = W[kt*32 + 16*(j/8) + 8*(lane/16) + j%8][nt*16 + lane%16]
__global__ void pack_kernel(const float* __restrict__ W, unsigned short* __restrict__ dst,
                            int NDIM, int total) {
    const int t = blockIdx.x * blockDim.x + threadIdx.x;
    if (t >= total) return;
    const int j      = t & 15;
    const int lane   = (t >> 4) & 31;
    const int blk    = t >> 9;
    const int ntiles = NDIM >> 4;
    const int kt     = blk / ntiles;
    const int nt     = blk - kt * ntiles;
    const int srcK   = (kt << 5) + ((j >> 3) << 4) + ((lane >> 4) << 3) + (j & 7);
    const int srcN   = (nt << 4) + (lane & 15);
    dst[t] = f2bf(W[srcK * NDIM + srcN]);
}

__global__ void count_kernel(const int* __restrict__ mapping, float* counts, int n) {
    const int i = blockIdx.x * blockDim.x + threadIdx.x;
    if (i < n) atomicAdd(&counts[mapping[i]], 1.0f);
}

__global__ void finalize_kernel(float* out, const float* __restrict__ counts) {
    const int i = blockIdx.x * blockDim.x + threadIdx.x;   // 4000*256 == 1,024,000 exactly
    if (i >= N_CG * FEAT * 4) return;
    const int cg = (i < N_CG * FEAT) ? (i >> 7) : ((i - N_CG * FEAT) / (3 * FEAT));
    out[i] *= 1.0f / fmaxf(counts[cg], 1.0f);
}

// ---------------- main edge kernel: 32 edges / block, 8 waves ----------------

__global__ __launch_bounds__(256) void edge_kernel(
    const float* __restrict__ s_i, const float* __restrict__ v_i,
    const float* __restrict__ r_iI,
    const float* __restrict__ b1,  const float* __restrict__ b2,
    const float* __restrict__ Wr,  const float* __restrict__ br,
    const int*   __restrict__ mapping,
    const unsigned short* __restrict__ W1p, const unsigned short* __restrict__ W2p,
    float* __restrict__ out_s, float* __restrict__ out_v)
{
    __shared__ unsigned short s_lds[EPB * FEAT];        // s_i tile, bf16        (8 KB)
    __shared__ unsigned short h_lds[EPB * FEAT];        // silu tile, bf16       (8 KB)
    __shared__ float          ws_lds[EPB * 3 * FEAT];   // (rbf@Wr+br)*env tile  (48 KB)
    __shared__ float          wr_lds[N_RBF * 3 * FEAT]; // Wr copy               (30 KB)
    __shared__ float          rbf_lds[EPB * N_RBF];
    __shared__ float          unit_lds[EPB][3];
    __shared__ float          env_lds[EPB];
    __shared__ int            cg_lds[EPB];

    const int tid   = threadIdx.x;
    const int wave  = tid >> 5;
    const int lane  = tid & 31;
    const int edge0 = blockIdx.x * EPB;

    // ---- kick off async Wr -> LDS copy (1920 x 16B), overlapped with staging below
    {
        const char* wr_g = (const char*)Wr;
        char* wr_l = (char*)wr_lds;
#pragma unroll
        for (int i = 0; i < 8; ++i) {
            const int idx = tid + (i << 8);
            if (idx < (N_RBF * 3 * FEAT * 4) / 16)      // 1920 chunks
                async_copy16(lds_addr_of(wr_l + idx * 16), wr_g + idx * 16);
        }
    }

    const int Nc    = lane & 15;
    const int Mbase = (lane >> 4) << 3;
    const int feat  = (wave << 4) + Nc;

    // Prefetch the v_i stream this lane will read at the end (pulls into L2 early).
    __builtin_prefetch(v_i + ((size_t)(edge0 + Mbase) * FEAT + feat) * 3, 0, 1);
    __builtin_prefetch(v_i + ((size_t)(edge0 + 16 + Mbase) * FEAT + feat) * 3, 0, 1);

    // ---- stage s_i tile -> LDS bf16 (32x128)
#pragma unroll
    for (int i = 0; i < 16; ++i) {
        const int idx = tid + (i << 8);
        const int e = idx >> 7, fcol = idx & 127;
        s_lds[idx] = f2bf(s_i[(size_t)(edge0 + e) * FEAT + fcol]);
    }

    // ---- per-edge geometry + RBF (one thread per edge)
    if (tid < EPB) {
        const float* rp = r_iI + (size_t)(edge0 + tid) * 3;
        const float x = rp[0], y = rp[1], z = rp[2];
        const float dist  = sqrtf(x * x + y * y + z * z + 3e-8f);
        const float inv_d = 1.0f / dist;
        unit_lds[tid][0] = x * inv_d;
        unit_lds[tid][1] = y * inv_d;
        unit_lds[tid][2] = z * inv_d;
        env_lds[tid] = (dist < CUTOFF_F)
                         ? 0.5f * (__cosf(PI_F * dist * (1.0f / CUTOFF_F)) + 1.0f) : 0.0f;
        cg_lds[tid] = mapping[edge0 + tid];
#pragma unroll
        for (int n = 0; n < N_RBF; ++n)
            rbf_lds[tid * N_RBF + n] =
                __sinf((float)(n + 1) * PI_F * dist * (1.0f / CUTOFF_F)) * inv_d;
    }

    wait_async0();          // Wr copy done (this wave's chunks)
    __syncthreads();        // all waves' LDS writes visible

    // ---- w_s = (rbf @ Wr + br) * env -> LDS (32 x 384), Wr from LDS, 48 items/thread
    for (int idx = tid; idx < EPB * 3 * FEAT; idx += 256) {
        const int e = idx / (3 * FEAT), col = idx - e * (3 * FEAT);
        float acc = br[col];
#pragma unroll
        for (int n = 0; n < N_RBF; ++n)
            acc += rbf_lds[e * N_RBF + n] * wr_lds[n * (3 * FEAT) + col];
        ws_lds[idx] = acc * env_lds[e];
    }

    // ---- GEMM1: h = silu(s @ W1 + b1); wave w owns cols [16w,16w+16), both M-halves
    {
        v8f acc[2] = {{}, {}};
#pragma unroll
        for (int kb = 0; kb < 4; ++kb) {
            const v16bf b  = load_b(W1p + (size_t)(((kb << 3) + wave) * 32 + lane) * 16);
            const v16bf a0 = gather_a(s_lds,            lane, kb);
            const v16bf a1 = gather_a(s_lds + 16 * FEAT, lane, kb);
            acc[0] = __builtin_amdgcn_wmma_f32_16x16x32_bf16(false, a0, false, b,
                                                             (short)0, acc[0], false, false);
            acc[1] = __builtin_amdgcn_wmma_f32_16x16x32_bf16(false, a1, false, b,
                                                             (short)0, acc[1], false, false);
        }
        const int   col = (wave << 4) + Nc;
        const float bs  = b1[col];
#pragma unroll
        for (int m2 = 0; m2 < 2; ++m2)
#pragma unroll
            for (int r = 0; r < 8; ++r) {
                const int M = (m2 << 4) + Mbase + r;
                float h = acc[m2][r] + bs;
                h = h * (1.0f / (1.0f + __expf(-h)));          // silu
                h_lds[M * FEAT + col] = f2bf(h);
            }
    }
    __syncthreads();

    // ---- GEMM2: phi = h @ W2 + b2; inv = phi * w_s.
    // Wave w takes N-tiles {w, w+8, w+16} == split0/1/2 for features [16w,16w+16).
    float sp0[2][8], sp2[2][8];
#pragma unroll
    for (int t = 0; t < 3; ++t) {
        const int nt = wave + (t << 3);
        v8f acc[2] = {{}, {}};
#pragma unroll
        for (int kb = 0; kb < 4; ++kb) {
            const v16bf b  = load_b(W2p + (size_t)((kb * 24 + nt) * 32 + lane) * 16);
            const v16bf a0 = gather_a(h_lds,            lane, kb);
            const v16bf a1 = gather_a(h_lds + 16 * FEAT, lane, kb);
            acc[0] = __builtin_amdgcn_wmma_f32_16x16x32_bf16(false, a0, false, b,
                                                             (short)0, acc[0], false, false);
            acc[1] = __builtin_amdgcn_wmma_f32_16x16x32_bf16(false, a1, false, b,
                                                             (short)0, acc[1], false, false);
        }
        const int   col = (nt << 4) + Nc;
        const float bs  = b2[col];
#pragma unroll
        for (int m2 = 0; m2 < 2; ++m2)
#pragma unroll
            for (int r = 0; r < 8; ++r) {
                const int M = (m2 << 4) + Mbase + r;
                const float val = (acc[m2][r] + bs) * ws_lds[M * (3 * FEAT) + col];
                if (t == 0)      sp0[m2][r] = val;                                // split_0
                else if (t == 2) sp2[m2][r] = val;                                // split_2
                else  atomicAdd(&out_s[(size_t)cg_lds[M] * FEAT + feat], val);    // split_1
            }
    }

    // ---- delta_v = split_2 * unit + split_0 * v_i -> scatter-add (v_i touched once)
#pragma unroll
    for (int m2 = 0; m2 < 2; ++m2)
#pragma unroll
        for (int r = 0; r < 8; ++r) {
            const int M  = (m2 << 4) + Mbase + r;
            const int cg = cg_lds[M];
            const float* vp = v_i + ((size_t)(edge0 + M) * FEAT + feat) * 3;
            float* op = out_v + ((size_t)cg * FEAT + feat) * 3;
            const float v0 = vp[0], v1 = vp[1], v2 = vp[2];
            atomicAdd(op + 0, sp2[m2][r] * unit_lds[M][0] + sp0[m2][r] * v0);
            atomicAdd(op + 1, sp2[m2][r] * unit_lds[M][1] + sp0[m2][r] * v1);
            atomicAdd(op + 2, sp2[m2][r] * unit_lds[M][2] + sp0[m2][r] * v2);
        }
}

// ---------------- launch ----------------

extern "C" void kernel_launch(void* const* d_in, const int* in_sizes, int n_in,
                              void* d_out, int out_size, void* d_ws, size_t ws_size,
                              hipStream_t stream) {
    (void)in_sizes; (void)n_in; (void)out_size; (void)ws_size;
    const float* s_i     = (const float*)d_in[0];
    const float* v_i     = (const float*)d_in[1];
    const float* r_iI    = (const float*)d_in[2];
    const float* W1      = (const float*)d_in[3];
    const float* b1      = (const float*)d_in[4];
    const float* W2      = (const float*)d_in[5];
    const float* b2      = (const float*)d_in[6];
    const float* Wr      = (const float*)d_in[7];
    const float* br      = (const float*)d_in[8];
    const int*   mapping = (const int*)d_in[9];

    float* out   = (float*)d_out;
    float* out_s = out;                       // (2000, 128)
    float* out_v = out + N_CG * FEAT;         // (2000, 128, 3)

    char* ws = (char*)d_ws;
    float*          counts = (float*)ws;                        // 8 KB region
    unsigned short* W1p    = (unsigned short*)(ws + 8192);      // 128*128 bf16 = 32 KB
    unsigned short* W2p    = (unsigned short*)(ws + 40960);     // 128*384 bf16 = 96 KB

    zero_kernel    <<<4096, 256, 0, stream>>>(out, N_CG * FEAT * 4, counts, N_CG);
    pack_kernel    <<<64,   256, 0, stream>>>(W1, W1p, FEAT,     FEAT * FEAT);
    pack_kernel    <<<192,  256, 0, stream>>>(W2, W2p, 3 * FEAT, FEAT * 3 * FEAT);
    count_kernel   <<<(N_EDGES + 255) / 256, 256, 0, stream>>>(mapping, counts, N_EDGES);
    edge_kernel    <<<N_EDGES / EPB, 256, 0, stream>>>(s_i, v_i, r_iI, b1, b2, Wr, br,
                                                       mapping, W1p, W2p, out_s, out_v);
    finalize_kernel<<<4000, 256, 0, stream>>>(out, counts);
}
